// PreNormTransformerLayer_86595130622110
// MI455X (gfx1250) — compile-verified, matrix operand order checked by reading, not run on hardware
//
// PreNormTransformerLayer for MI455X (gfx1250), wave32 + WMMA bf16 + async LDS.
//
// Pipeline (all on `stream`, all scratch in d_ws):
//  1. transpose-convert 6 weight matrices fp32 -> bf16 [out][in]
//  2. LN1(x) -> xn bf16
//  3. Q,K = xn@W{q,k} -> [B,H,L,HD] bf16 ; V -> head-transposed [B,H,HD,L] bf16   (WMMA)
//  4. pair bias: bias[b,h,q,k] = pair[b,q,k,:]@Wp[:,h] + bp  (fp32 VALU, 4 flop/byte -> mem bound)
//  5. scores: S = scale * q.k^T + bias (fp32, in-place over bias buffer)          (WMMA)
//  6. masked softmax rows -> P bf16
//  7. attn_out = P@V  -> [B,L,D] bf16                                             (WMMA)
//  8. x1 = x + attn_out@Wo + bo (fp32)                                            (WMMA)
//  9. LN2(x1) -> xn2 bf16
// 10. hbuf = gelu(xn2@W1 + b1) bf16                                               (WMMA)
// 11. out = x1 + hbuf@W2 + b2 (fp32 -> d_out)                                     (WMMA)
//
// k_gemm stages the shared B-tile through LDS with GLOBAL_LOAD_ASYNC_TO_LDS_B128
// (double-buffered, s_wait_asynccnt + barrier), 4x reuse across the block's waves.

#include <hip/hip_runtime.h>
#include <hip/hip_bf16.h>
#include <math.h>

#define DEV_INLINE __device__ __forceinline__

typedef __attribute__((ext_vector_type(16))) __bf16 v16bf;
typedef __attribute__((ext_vector_type(8)))  __bf16 v8bf;
typedef __attribute__((ext_vector_type(8)))  float  v8f;

constexpr int kB = 2, kL = 512, kD = 256, kH = 8, kF = 1024;
constexpr int kHD = kD / kH;      // 32 == one bf16 WMMA K-step
constexpr int kBL = kB * kL;      // 1024
constexpr float kEPS   = 1e-5f;
constexpr float kSCALE = 0.17677669529663687f;  // 1/sqrt(32)

// ---------------------------------------------------------------------------
// Fragment loader. ISA 16-bit A layout (CDNA5 ISA §7.12.2):
// lane l (half = l>>4, row = l&15), element e: K = 16*(e/8) + 8*half + (e%8).
// => two contiguous 8-element (16B) chunks at k0 + 8*half and k0 + 16 + 8*half.
// Works for generic (global) and LDS pointers alike (ds_load_b128 for LDS).
// ---------------------------------------------------------------------------
DEV_INLINE v16bf load_frag(const __bf16* row, int k0, int half) {
  const __bf16* p = row + k0 + 8 * half;
  v8bf lo = *reinterpret_cast<const v8bf*>(p);        // K = 8*half .. +7
  v8bf hi = *reinterpret_cast<const v8bf*>(p + 16);   // K = 16+8*half .. +7
  v16bf f;
#pragma unroll
  for (int i = 0; i < 8; ++i) { f[i] = lo[i]; f[i + 8] = hi[i]; }
  return f;
}

DEV_INLINE v8f wmma_bf16(v16bf a, v16bf b, v8f c) {
  // (neg_a, A, neg_b, B, c_mod, C, reuse_a, reuse_b)
  return __builtin_amdgcn_wmma_f32_16x16x32_bf16(false, a, false, b,
                                                 (short)0, c, false, false);
}

// Async global->LDS 16B copy (per-lane), tracked by ASYNCcnt.
DEV_INLINE void async_copy_b128(unsigned lds_byte_addr, const void* gaddr) {
  asm volatile("global_load_async_to_lds_b128 %0, %1, off"
               :: "v"(lds_byte_addr), "v"(gaddr) : "memory");
}
DEV_INLINE void wait_asynccnt0() {
  asm volatile("s_wait_asynccnt 0x0" ::: "memory");
}

// ---------------------------------------------------------------------------
// Weight fp32 [rows(in) x cols(out)] -> bf16 transposed [cols][rows]
// ---------------------------------------------------------------------------
__global__ void k_cvt_transpose(const float* __restrict__ src,
                                __bf16* __restrict__ dst, int rows, int cols) {
  int idx = blockIdx.x * blockDim.x + threadIdx.x;
  if (idx >= rows * cols) return;
  int r = idx / cols, c = idx % cols;
  dst[c * rows + r] = (__bf16)src[idx];
}

// ---------------------------------------------------------------------------
// LayerNorm over D=256, one wave per row, 8 elems per lane, shfl_xor reduce.
// ---------------------------------------------------------------------------
__global__ void k_layernorm_bf16(const float* __restrict__ x,
                                 const float* __restrict__ w,
                                 const float* __restrict__ b,
                                 __bf16* __restrict__ out) {
  int wave = threadIdx.x >> 5;
  int lane = threadIdx.x & 31;
  int row  = blockIdx.x * (blockDim.x >> 5) + wave;
  if (row >= kBL) return;
  const float* xr = x + row * kD;
  float v[8];
  float s = 0.f;
#pragma unroll
  for (int j = 0; j < 8; ++j) { v[j] = xr[lane + 32 * j]; s += v[j]; }
#pragma unroll
  for (int off = 16; off; off >>= 1) s += __shfl_xor(s, off, 32);
  float mean = s * (1.0f / kD);
  float sv = 0.f;
#pragma unroll
  for (int j = 0; j < 8; ++j) { float d = v[j] - mean; sv += d * d; }
#pragma unroll
  for (int off = 16; off; off >>= 1) sv += __shfl_xor(sv, off, 32);
  float inv = rsqrtf(sv * (1.0f / kD) + kEPS);
#pragma unroll
  for (int j = 0; j < 8; ++j) {
    int d = lane + 32 * j;
    out[row * kD + d] = (__bf16)((v[j] - mean) * inv * w[d] + b[d]);
  }
}

// ---------------------------------------------------------------------------
// WMMA GEMM with async-LDS staged B operand.
// Block = 128 threads = 4 waves = 4 stacked M-tiles x 1 N-tile, so all waves
// share one 16x32 B tile per K-step: double-buffered in LDS (2 x 1KB), staged
// by lanes 0..63 (one async b128 each), consumed via ds_load_b128 fragments.
// Requires M % 64 == 0 (true for all uses; grids are exact, no early-outs).
// Epilogue modes:
//   0: bf16 row-major     1: q/k headed [B,H,L,HD]   2: v transposed [B,H,HD,L]
//   3: gelu -> bf16       4: fp32 += residual
// ---------------------------------------------------------------------------
__global__ void k_gemm(const __bf16* __restrict__ A, int lda,
                       const __bf16* __restrict__ Bt, int ldb,
                       const float* __restrict__ bias,
                       const float* __restrict__ resid,
                       void* __restrict__ out,
                       int M, int N, int K, int mode) {
  __shared__ __align__(16) __bf16 sB[2][16 * 32];   // 2 x 1KB double buffer

  int tid  = threadIdx.x;
  int lane = tid & 31;
  int half = lane >> 4;
  int ln16 = lane & 15;
  int tn   = blockIdx.x % (N >> 4);
  int tm   = (blockIdx.x / (N >> 4)) * 4 + (tid >> 5);

  const __bf16* arow  = A  + (tm * 16 + ln16) * lda;
  const __bf16* bbase = Bt + (size_t)(tn * 16) * ldb;  // shared across block
  unsigned lds0 = (unsigned)(size_t)(&sB[0][0]);       // low 32 bits = LDS offset

  // stage one 16x32 bf16 B tile (k-slice at k0) into buffer `buf`
  auto stage = [&](int buf, int k0) {
    if (tid < 64) {                                    // waves 0,1: full EXEC
      int row = tid >> 2, ch = tid & 3;                // 16 rows x 4 chunks
      const __bf16* g = bbase + row * ldb + k0 + ch * 8;
      async_copy_b128(lds0 + (unsigned)(buf * 1024 + row * 64 + ch * 16), g);
    }
  };

  stage(0, 0);
  v8f acc = {};
  for (int k0 = 0, it = 0; k0 < K; k0 += 32, ++it) {
    int cur = it & 1;
    wait_asynccnt0();          // staging waves wait for their async copies
    __syncthreads();           // staged tile visible; prior reads of buf cur done
    if (k0 + 32 < K) {
      stage(cur ^ 1, k0 + 32);
      __builtin_prefetch(arow + k0 + 32, 0, 3);        // global_prefetch_b8
    }
    v16bf a = load_frag(arow, k0, half);               // global b128 x2
    v16bf b = load_frag(&sB[cur][ln16 * 32], 0, half); // ds_load_b128 x2
    acc = wmma_bf16(a, b, acc);
  }

  int n = tn * 16 + ln16;
  float bn = bias ? bias[n] : 0.f;
#pragma unroll
  for (int r = 0; r < 8; ++r) {
    int m = tm * 16 + r + 8 * half;                    // ISA C-layout
    float c = acc[r] + bn;
    if (mode == 0) {
      ((__bf16*)out)[m * N + n] = (__bf16)c;
    } else if (mode == 1) {                            // [B,H,L,HD]
      int bb = m / kL, qp = m % kL, hh = n / kHD, dd = n % kHD;
      ((__bf16*)out)[((bb * kH + hh) * kL + qp) * kHD + dd] = (__bf16)c;
    } else if (mode == 2) {                            // [B,H,HD,L]
      int bb = m / kL, qp = m % kL, hh = n / kHD, dd = n % kHD;
      ((__bf16*)out)[((bb * kH + hh) * kHD + dd) * kL + qp] = (__bf16)c;
    } else if (mode == 3) {                            // exact gelu
      float g = 0.5f * c * (1.0f + erff(c * 0.70710678118f));
      ((__bf16*)out)[m * N + n] = (__bf16)g;
    } else {                                           // fp32 + residual
      float rr = resid ? resid[m * N + n] : 0.f;
      ((float*)out)[m * N + n] = c + rr;
    }
  }
}

// ---------------------------------------------------------------------------
// Pair bias (memory-bound, 4 flop/byte -> fp32 VALU). One wave per (b,q,k):
// 32 lanes x 8 d's, Wp cached in LDS, shfl_xor tree reduce per head.
// ---------------------------------------------------------------------------
__global__ void k_pair_bias(const float* __restrict__ pair,
                            const float* __restrict__ Wp,
                            const float* __restrict__ bp,
                            float* __restrict__ Sbias) {
  __shared__ float sWp[kD * kH];                       // 8 KB
  for (int i = threadIdx.x; i < kD * kH; i += blockDim.x) sWp[i] = Wp[i];
  __syncthreads();
  int lane = threadIdx.x & 31;
  int wave = threadIdx.x >> 5;
  int pairIdx = blockIdx.x * (blockDim.x >> 5) + wave; // < B*L*L = 524288
  if (pairIdx >= kB * kL * kL) return;
  int bq = pairIdx / kL;                               // b*L + q
  int kpos = pairIdx % kL;
  const float* pv = pair + (size_t)pairIdx * kD;
  float acc[kH] = {};
#pragma unroll
  for (int j = 0; j < 8; ++j) {
    float xv = pv[lane + 32 * j];                      // coalesced 128B/wave
    const float* wr = &sWp[(lane + 32 * j) * kH];
#pragma unroll
    for (int h = 0; h < kH; ++h) acc[h] += xv * wr[h];
  }
#pragma unroll
  for (int h = 0; h < kH; ++h) {
#pragma unroll
    for (int off = 16; off; off >>= 1) acc[h] += __shfl_xor(acc[h], off, 32);
  }
  if (lane == 0) {
    int b = bq / kL, q = bq % kL;
#pragma unroll
    for (int h = 0; h < kH; ++h)
      Sbias[((b * kH + h) * kL + q) * kL + kpos] = acc[h] + bp[h];
  }
}

// ---------------------------------------------------------------------------
// Scores: S = scale*q.k^T + bias (in place). HD=32 -> single WMMA per tile.
// ---------------------------------------------------------------------------
__global__ void k_scores(const __bf16* __restrict__ q,
                         const __bf16* __restrict__ k,
                         float* __restrict__ S) {
  int lane = threadIdx.x & 31, half = lane >> 4, ln16 = lane & 15;
  int gid = blockIdx.x * (blockDim.x >> 5) + (threadIdx.x >> 5);
  int bh  = gid >> 10;                                 // 32*32 tiles per (b,h)
  int rem = gid & 1023;
  int tq = rem >> 5, tk = rem & 31;
  const __bf16* qrow = q + (bh * kL + tq * 16 + ln16) * kHD;
  const __bf16* krow = k + (bh * kL + tk * 16 + ln16) * kHD;
  v8f acc = {};
  acc = wmma_bf16(load_frag(qrow, 0, half), load_frag(krow, 0, half), acc);
#pragma unroll
  for (int r = 0; r < 8; ++r) {
    int m = tq * 16 + r + 8 * half;
    int n = tk * 16 + ln16;
    int idx = (bh * kL + m) * kL + n;
    S[idx] = acc[r] * kSCALE + S[idx];
  }
}

// ---------------------------------------------------------------------------
// Masked softmax over rows of length L=512; 256 threads, 2 elems/thread.
// ---------------------------------------------------------------------------
__global__ void k_softmax(const float* __restrict__ S, __bf16* __restrict__ P,
                          const unsigned char* __restrict__ mask) {
  int row = blockIdx.x;                                // [0, B*H*L)
  int b = row / (kH * kL);
  int base = row * kL;
  const unsigned char* mrow = mask + b * kL;
  int t = threadIdx.x;
  float v0 = mrow[t]       ? S[base + t]       : -INFINITY;
  float v1 = mrow[t + 256] ? S[base + t + 256] : -INFINITY;
  __shared__ float red[8];
  float mx = fmaxf(v0, v1);
#pragma unroll
  for (int off = 16; off; off >>= 1) mx = fmaxf(mx, __shfl_xor(mx, off, 32));
  if ((t & 31) == 0) red[t >> 5] = mx;
  __syncthreads();
  float rmax = red[0];
#pragma unroll
  for (int i = 1; i < 8; ++i) rmax = fmaxf(rmax, red[i]);
  __syncthreads();
  float e0 = (v0 > -INFINITY) ? __expf(v0 - rmax) : 0.f;
  float e1 = (v1 > -INFINITY) ? __expf(v1 - rmax) : 0.f;
  float s = e0 + e1;
#pragma unroll
  for (int off = 16; off; off >>= 1) s += __shfl_xor(s, off, 32);
  if ((t & 31) == 0) red[t >> 5] = s;
  __syncthreads();
  float rsum = red[0];
#pragma unroll
  for (int i = 1; i < 8; ++i) rsum += red[i];
  float inv = 1.0f / rsum;
  P[base + t]       = (__bf16)(e0 * inv);
  P[base + t + 256] = (__bf16)(e1 * inv);
}

// ---------------------------------------------------------------------------
// attn_out = P @ V : per (b,h), A=P [L x L] bf16, Bt=Vt [HD][L] bf16.
// ---------------------------------------------------------------------------
__global__ void k_attnv(const __bf16* __restrict__ P,
                        const __bf16* __restrict__ Vt,
                        __bf16* __restrict__ Aout) {
  int lane = threadIdx.x & 31, half = lane >> 4, ln16 = lane & 15;
  int gid = blockIdx.x * (blockDim.x >> 5) + (threadIdx.x >> 5);
  int bh  = gid / 64;                                  // 32 q-tiles * 2 d-tiles
  int rem = gid % 64;
  int tq = rem >> 1, td = rem & 1;
  const __bf16* prow = P  + (bh * kL + tq * 16 + ln16) * kL;
  const __bf16* vrow = Vt + (bh * kHD + td * 16 + ln16) * kL;
  v8f acc = {};
  for (int k0 = 0; k0 < kL; k0 += 32) {
    if (k0 + 32 < kL) {
      __builtin_prefetch(prow + k0 + 32, 0, 3);
      __builtin_prefetch(vrow + k0 + 32, 0, 3);
    }
    acc = wmma_bf16(load_frag(prow, k0, half), load_frag(vrow, k0, half), acc);
  }
  int b = bh / kH, h = bh % kH;
#pragma unroll
  for (int r = 0; r < 8; ++r) {
    int qp = tq * 16 + r + 8 * half;
    int dd = td * 16 + ln16;
    Aout[(b * kL + qp) * kD + h * kHD + dd] = (__bf16)acc[r];
  }
}

// ---------------------------------------------------------------------------
extern "C" void kernel_launch(void* const* d_in, const int* in_sizes, int n_in,
                              void* d_out, int out_size, void* d_ws, size_t ws_size,
                              hipStream_t stream) {
  const float* x    = (const float*)d_in[0];
  const float* pair = (const float*)d_in[1];
  const unsigned char* mask = (const unsigned char*)d_in[2];  // jnp bool (1B)
  const float* ln1w = (const float*)d_in[3];
  const float* ln1b = (const float*)d_in[4];
  const float* ln2w = (const float*)d_in[5];
  const float* ln2b = (const float*)d_in[6];
  const float* Wq = (const float*)d_in[7];  const float* bq = (const float*)d_in[8];
  const float* Wk = (const float*)d_in[9];  const float* bk = (const float*)d_in[10];
  const float* Wv = (const float*)d_in[11]; const float* bv = (const float*)d_in[12];
  const float* Wo = (const float*)d_in[13]; const float* bo = (const float*)d_in[14];
  const float* Wp = (const float*)d_in[15]; const float* bp = (const float*)d_in[16];
  const float* W1 = (const float*)d_in[17]; const float* b1 = (const float*)d_in[18];
  const float* W2 = (const float*)d_in[19]; const float* b2 = (const float*)d_in[20];
  float* out = (float*)d_out;
  (void)in_sizes; (void)n_in; (void)out_size; (void)ws_size;

  // ---- carve workspace (256B aligned chunks), ~34 MB total ----
  char* p = (char*)d_ws;
  auto carve = [&](size_t bytes) {
    char* r = p; p += (bytes + 255) & ~(size_t)255; return (void*)r;
  };
  __bf16* wq_t = (__bf16*)carve((size_t)kD * kD * 2);
  __bf16* wk_t = (__bf16*)carve((size_t)kD * kD * 2);
  __bf16* wv_t = (__bf16*)carve((size_t)kD * kD * 2);
  __bf16* wo_t = (__bf16*)carve((size_t)kD * kD * 2);
  __bf16* w1_t = (__bf16*)carve((size_t)kD * kF * 2);
  __bf16* w2_t = (__bf16*)carve((size_t)kF * kD * 2);
  __bf16* xn   = (__bf16*)carve((size_t)kBL * kD * 2);
  __bf16* qb   = (__bf16*)carve((size_t)kBL * kD * 2);    // [B,H,L,HD]
  __bf16* kb   = (__bf16*)carve((size_t)kBL * kD * 2);    // [B,H,L,HD]
  __bf16* vt   = (__bf16*)carve((size_t)kBL * kD * 2);    // [B,H,HD,L]
  float*  Sbuf = (float*) carve((size_t)kB * kH * kL * kL * 4);  // bias then scores
  __bf16* Pbuf = (__bf16*)carve((size_t)kB * kH * kL * kL * 2);
  __bf16* aout = (__bf16*)carve((size_t)kBL * kD * 2);
  float*  x1   = (float*) carve((size_t)kBL * kD * 4);
  __bf16* xn2  = (__bf16*)carve((size_t)kBL * kD * 2);
  __bf16* hbuf = (__bf16*)carve((size_t)kBL * kF * 2);

  // 1) weight convert/transpose
  auto cvt = [&](const float* s, __bf16* d, int r, int c) {
    int n = r * c;
    k_cvt_transpose<<<(n + 255) / 256, 256, 0, stream>>>(s, d, r, c);
  };
  cvt(Wq, wq_t, kD, kD);  cvt(Wk, wk_t, kD, kD);
  cvt(Wv, wv_t, kD, kD);  cvt(Wo, wo_t, kD, kD);
  cvt(W1, w1_t, kD, kF);  cvt(W2, w2_t, kF, kD);

  // 2) LN1
  k_layernorm_bf16<<<kBL / 8, 256, 0, stream>>>(x, ln1w, ln1b, xn);

  // 3) QKV projections (WMMA + async-LDS B). grid = (M/64)*(N/16) = 16*16
  const int gqkv = (kBL / 64) * (kD / 16);
  k_gemm<<<gqkv, 128, 0, stream>>>(xn, kD, wq_t, kD, bq, nullptr, qb, kBL, kD, kD, 1);
  k_gemm<<<gqkv, 128, 0, stream>>>(xn, kD, wk_t, kD, bk, nullptr, kb, kBL, kD, kD, 1);
  k_gemm<<<gqkv, 128, 0, stream>>>(xn, kD, wv_t, kD, bv, nullptr, vt, kBL, kD, kD, 2);

  // 4) pair bias (memory-bound fp32 path): one wave per (b,q,k)
  k_pair_bias<<<(kB * kL * kL) / 8, 256, 0, stream>>>(pair, Wp, bp, Sbuf);

  // 5) scores (WMMA, K=HD=32): B*H*32*32 = 16384 waves / 4
  k_scores<<<(kB * kH * (kL / 16) * (kL / 16)) / 4, 128, 0, stream>>>(qb, kb, Sbuf);

  // 6) softmax -> P bf16
  k_softmax<<<kB * kH * kL, 256, 0, stream>>>(Sbuf, Pbuf, mask);

  // 7) attn_out = P@V (WMMA): B*H*32*2 = 1024 waves / 4
  k_attnv<<<(kB * kH * (kL / 16) * (kHD / 16)) / 4, 128, 0, stream>>>(Pbuf, vt, aout);

  // 8) x1 = x + attn_out@Wo + bo
  k_gemm<<<gqkv, 128, 0, stream>>>(aout, kD, wo_t, kD, bo, x, x1, kBL, kD, kD, 4);

  // 9) LN2
  k_layernorm_bf16<<<kBL / 8, 256, 0, stream>>>(x1, ln2w, ln2b, xn2);

  // 10) FFN up + exact gelu: grid = 16 * 64
  k_gemm<<<(kBL / 64) * (kF / 16), 128, 0, stream>>>(xn2, kD, w1_t, kD, b1,
                                                     nullptr, hbuf, kBL, kF, kD, 3);

  // 11) FFN down + residual -> d_out (fp32)
  k_gemm<<<gqkv, 128, 0, stream>>>(hbuf, kF, w2_t, kF, b2, x1, out, kBL, kD, kF, 4);
}
